// MyAutoCorrelation_46359876993771
// MI455X (gfx1250) — matrix operand order repaired and checked by reading, not compiled, not run on hardware
//
#include <hip/hip_runtime.h>
#include <hip/hip_bf16.h>
#include <math.h>

#define B_DIM 16
#define T_DIM 4096
#define E_DIM 512
#define TOPK  16
#define NTILE (T_DIM / 16)      // 256 column tiles
#define IBLKS (T_DIM / 128)     // 32 row blocks (128 rows each, 8 waves x 16)

typedef __attribute__((ext_vector_type(16))) __bf16        v16bf;
typedef __attribute__((ext_vector_type(8)))  float         v8f;
typedef __attribute__((ext_vector_type(8)))  unsigned short u16x8;
typedef __attribute__((ext_vector_type(4)))  unsigned short u16x4;

union FragAB { u16x8 h[2]; v16bf v; };

// ---------------------------------------------------------------------------
// Phase 0: f32 -> bf16 (RNE) conversion of Q and K into workspace
// ---------------------------------------------------------------------------
__device__ __forceinline__ unsigned short f32_to_bf16(float f) {
    unsigned int x = __float_as_uint(f);
    unsigned int r = x + 0x7FFFu + ((x >> 16) & 1u);
    return (unsigned short)(r >> 16);
}

__global__ __launch_bounds__(256)
void ac_convert_bf16(const float* __restrict__ Q, const float* __restrict__ K,
                     unsigned short* __restrict__ Qb, unsigned short* __restrict__ Kb,
                     int n4) {
    int i = blockIdx.x * 256 + threadIdx.x;
    if (i >= n4) return;
    const float4 q = ((const float4*)Q)[i];
    const float4 k = ((const float4*)K)[i];
    u16x4 qo, ko;
    qo.x = f32_to_bf16(q.x); qo.y = f32_to_bf16(q.y);
    qo.z = f32_to_bf16(q.z); qo.w = f32_to_bf16(q.w);
    ko.x = f32_to_bf16(k.x); ko.y = f32_to_bf16(k.y);
    ko.z = f32_to_bf16(k.z); ko.w = f32_to_bf16(k.w);
    ((u16x4*)Qb)[i] = qo;
    ((u16x4*)Kb)[i] = ko;
}

// ---------------------------------------------------------------------------
// Phase 1: WMMA Gram-matrix with wrapped-diagonal LDS accumulation
// grid = (IBLKS, B), block = 256 (8 waves). Dynamic LDS:
//   [0, 128KB)   : 8 per-wave diagonal accumulators (4096 f32 each)
//   [128,160KB)  : double-buffered 16x512 bf16 K tile
// P[b][iblk][d] = sum over this block's 128 Q-rows of G along diagonal d
// ---------------------------------------------------------------------------
__global__ __launch_bounds__(256, 1)
void ac_gram_diag(const unsigned short* __restrict__ Qb,
                  const unsigned short* __restrict__ Kb,
                  float* __restrict__ P) {
    extern __shared__ char smem[];
    float*          s_acc = (float*)smem;                           // 8 * 4096
    unsigned short* s_B   = (unsigned short*)(smem + 8 * 4096 * 4); // 2 * 16 * 512

    const int b    = blockIdx.y;
    const int iblk = blockIdx.x;
    const int tid  = threadIdx.x;
    const int wave = tid >> 5;
    const int lane = tid & 31;
    const int half = lane >> 4;
    const int l16  = lane & 15;

    for (int i = tid; i < 8 * T_DIM; i += 256) s_acc[i] = 0.0f;
    float* accW = s_acc + wave * T_DIM;

    // Register-cache this wave's 16x512 A tile as 16 WMMA fragments.
    // A layout (16-bit, 16x32): lane m=l16; half=0 -> K 0..7 & 16..23,
    // half=1 -> K 8..15 & 24..31  => two contiguous 16B runs per fragment.
    const int n0 = iblk * 128 + wave * 16;
    const unsigned short* qrow = Qb + ((size_t)(b * T_DIM + n0 + l16)) * E_DIM;
    FragAB afrag[16];
#pragma unroll
    for (int kc = 0; kc < 16; ++kc) {
        const unsigned short* p = qrow + kc * 32 + half * 8;
        afrag[kc].h[0] = *(const u16x8*)(p);
        afrag[kc].h[1] = *(const u16x8*)(p + 16);
    }

    // cooperative stage of K tile j into buffer buf (16 rows x 512 bf16)
    auto stage = [&](int j, int buf) {
        const int r = tid >> 4;          // row 0..15
        const int c = tid & 15;          // 32-element chunk 0..15
        const unsigned short* src =
            Kb + ((size_t)(b * T_DIM + j * 16 + r)) * E_DIM + c * 32;
        unsigned short* dst = s_B + buf * (16 * E_DIM) + r * E_DIM + c * 32;
        *(u16x8*)(dst)      = *(const u16x8*)(src);
        *(u16x8*)(dst + 8)  = *(const u16x8*)(src + 8);
        *(u16x8*)(dst + 16) = *(const u16x8*)(src + 16);
        *(u16x8*)(dst + 24) = *(const u16x8*)(src + 24);
        // pull the tile two steps ahead into L2 (global_prefetch_b8)
        __builtin_prefetch(src + 16 * E_DIM, 0, 1);
    };

    stage(0, 0);
    __syncthreads();

    for (int j = 0; j < NTILE; ++j) {
        if (j + 1 < NTILE) stage(j + 1, (j + 1) & 1);

        const unsigned short* Bc = s_B + (j & 1) * (16 * E_DIM);

        // B layout (16-bit, 32x16): lane N=l16; half=0 -> K 0..7 & 16..23, half=1 -> K 8..15 & 24..31
        auto loadB = [&](FragAB& f, int kc) {
            const unsigned short* p = Bc + l16 * E_DIM + kc * 32 + half * 8;
            f.h[0] = *(const u16x8*)(p);
            f.h[1] = *(const u16x8*)(p + 16);
        };

        // Two independent accumulator chains (even/odd kc) + depth-2
        // software pipeline of B fragments: keeps 4 ds_load_b128 in flight
        // while both WMMA chains issue, hiding LDS latency.
        v8f acc0 = {}, acc1 = {};
        FragAB bf[4];
        loadB(bf[0], 0);
        loadB(bf[1], 1);
#pragma unroll
        for (int kc = 0; kc < 16; kc += 2) {
            const int cur = (kc >> 1) & 1;
            const int nxt = cur ^ 1;
            if (kc + 2 < 16) {
                loadB(bf[nxt * 2 + 0], kc + 2);
                loadB(bf[nxt * 2 + 1], kc + 3);
            }
            acc0 = __builtin_amdgcn_wmma_f32_16x16x32_bf16(
                false, afrag[kc + 0].v, false, bf[cur * 2 + 0].v,
                (short)0, acc0, false, false);
            acc1 = __builtin_amdgcn_wmma_f32_16x16x32_bf16(
                false, afrag[kc + 1].v, false, bf[cur * 2 + 1].v,
                (short)0, acc1, false, false);
        }
        const v8f acc = acc0 + acc1;

        // scatter tile into wrapped diagonals: d = (m - n) mod T
        // D layout: VGPR r -> M = r + 8*half, lane -> N = l16
        const int base = n0 - j * 16 + 8 * half - l16 + 2 * T_DIM;
#pragma unroll
        for (int r = 0; r < 8; ++r)
            atomicAdd(&accW[(base + r) & (T_DIM - 1)], acc[r]);

        __syncthreads();
    }

    // reduce the 8 per-wave accumulators -> partial row
    for (int d = tid; d < T_DIM; d += 256) {
        float s = 0.0f;
#pragma unroll
        for (int wv = 0; wv < 8; ++wv) s += s_acc[wv * T_DIM + d];
        P[((size_t)b * IBLKS + iblk) * T_DIM + d] = s;
    }
}

// ---------------------------------------------------------------------------
// Phase 2: reduce partials -> mean_corr[b][d], global_mean[d]
// ---------------------------------------------------------------------------
__global__ __launch_bounds__(256)
void ac_reduce(const float* __restrict__ P, float* __restrict__ mean_corr,
               float* __restrict__ global_mean) {
    int d = blockIdx.x * 256 + threadIdx.x;
    if (d >= T_DIM) return;
    float g = 0.0f;
    for (int b = 0; b < B_DIM; ++b) {
        float s = 0.0f;
        for (int ib = 0; ib < IBLKS; ++ib)
            s += P[((size_t)b * IBLKS + ib) * T_DIM + d];
        s *= (1.0f / (float)E_DIM);
        mean_corr[b * T_DIM + d] = s;
        g += s;
    }
    global_mean[d] = g * (1.0f / (float)B_DIM);
}

// ---------------------------------------------------------------------------
// Phase 3: top-16 delays of global_mean, then per-batch softmax weights
// ---------------------------------------------------------------------------
__global__ __launch_bounds__(256)
void ac_topk_softmax(const float* __restrict__ mean_corr,
                     const float* __restrict__ global_mean,
                     int* __restrict__ delays, float* __restrict__ w) {
    __shared__ float sv[T_DIM];
    __shared__ float rv[256];
    __shared__ int   ri[256];
    __shared__ int   sdel[TOPK];
    const int tid = threadIdx.x;

    for (int d = tid; d < T_DIM; d += 256) sv[d] = global_mean[d];
    __syncthreads();

    for (int k = 0; k < TOPK; ++k) {
        float best = -INFINITY; int bi = T_DIM;
        for (int d = tid; d < T_DIM; d += 256) {
            float v = sv[d];
            if (v > best) { best = v; bi = d; }
        }
        rv[tid] = best; ri[tid] = bi;
        __syncthreads();
        for (int s = 128; s > 0; s >>= 1) {
            if (tid < s) {
                float v2 = rv[tid + s]; int i2 = ri[tid + s];
                if (v2 > rv[tid] || (v2 == rv[tid] && i2 < ri[tid])) {
                    rv[tid] = v2; ri[tid] = i2;
                }
            }
            __syncthreads();
        }
        if (tid == 0) { sdel[k] = ri[0]; delays[k] = ri[0]; sv[ri[0]] = -INFINITY; }
        __syncthreads();
    }

    if (tid < B_DIM) {
        float vals[TOPK]; float mx = -INFINITY;
        for (int k = 0; k < TOPK; ++k) {
            vals[k] = mean_corr[tid * T_DIM + sdel[k]];
            mx = fmaxf(mx, vals[k]);
        }
        float sum = 0.0f;
        for (int k = 0; k < TOPK; ++k) { vals[k] = __expf(vals[k] - mx); sum += vals[k]; }
        float inv = 1.0f / sum;
        for (int k = 0; k < TOPK; ++k) w[tid * TOPK + k] = vals[k] * inv;
    }
}

// ---------------------------------------------------------------------------
// Phase 4: out[b,t,:] = sum_k w[b,k] * V[b,(t+delay_k)%T,:]
// grid = (T, B), block = 256 (float2 per thread covers E=512)
// ---------------------------------------------------------------------------
__global__ __launch_bounds__(256)
void ac_gather(const float* __restrict__ V, const int* __restrict__ delays,
               const float* __restrict__ w, float* __restrict__ out) {
    __shared__ int   sd[TOPK];
    __shared__ float sw[TOPK];
    const int b = blockIdx.y, t = blockIdx.x, tid = threadIdx.x;
    if (tid < TOPK) { sd[tid] = delays[tid]; sw[tid] = w[b * TOPK + tid]; }
    __syncthreads();

    const int e = tid * 2;
    float2 acc = make_float2(0.0f, 0.0f);
#pragma unroll 4
    for (int k = 0; k < TOPK; ++k) {
        const int src = (t + sd[k]) & (T_DIM - 1);
        const float2 v = *(const float2*)(V + ((size_t)b * T_DIM + src) * E_DIM + e);
        acc.x += sw[k] * v.x;
        acc.y += sw[k] * v.y;
    }
    *(float2*)(out + ((size_t)b * T_DIM + t) * E_DIM + e) = acc;
}

// ---------------------------------------------------------------------------
extern "C" void kernel_launch(void* const* d_in, const int* in_sizes, int n_in,
                              void* d_out, int out_size, void* d_ws, size_t ws_size,
                              hipStream_t stream) {
    const float* Q = (const float*)d_in[0];
    const float* K = (const float*)d_in[1];
    const float* V = (const float*)d_in[2];
    float* out = (float*)d_out;

    const size_t nQK = (size_t)B_DIM * T_DIM * E_DIM;   // 33.55M elements

    // workspace layout (256B aligned regions)
    char* ws = (char*)d_ws;
    size_t off = 0;
    auto alloc = [&](size_t bytes) {
        void* p = ws + off;
        off += (bytes + 255) & ~(size_t)255;
        return p;
    };
    unsigned short* Qb   = (unsigned short*)alloc(nQK * sizeof(unsigned short));
    unsigned short* Kb   = (unsigned short*)alloc(nQK * sizeof(unsigned short));
    float* P             = (float*)alloc((size_t)B_DIM * IBLKS * T_DIM * sizeof(float));
    float* mean_corr     = (float*)alloc((size_t)B_DIM * T_DIM * sizeof(float));
    float* global_mean   = (float*)alloc((size_t)T_DIM * sizeof(float));
    int*   delays        = (int*)alloc(TOPK * sizeof(int));
    float* w             = (float*)alloc((size_t)B_DIM * TOPK * sizeof(float));
    (void)ws_size;

    // Phase 1 needs 160KB of LDS (CDNA5 has 320KB/WGP); raise the cap.
    const int ldsBytes = (8 * T_DIM * 4) + (2 * 16 * E_DIM * 2); // 128KB + 32KB
    (void)hipFuncSetAttribute((const void*)ac_gram_diag,
                              hipFuncAttributeMaxDynamicSharedMemorySize, ldsBytes);

    const int n4 = (int)(nQK / 4);
    ac_convert_bf16<<<(n4 + 255) / 256, 256, 0, stream>>>(Q, K, Qb, Kb, n4);

    ac_gram_diag<<<dim3(IBLKS, B_DIM), 256, ldsBytes, stream>>>(Qb, Kb, P);

    ac_reduce<<<T_DIM / 256, 256, 0, stream>>>(P, mean_corr, global_mean);

    ac_topk_softmax<<<1, 256, 0, stream>>>(mean_corr, global_mean, delays, w);

    ac_gather<<<dim3(T_DIM, B_DIM), 256, 0, stream>>>(V, delays, w, out);
}
